// MultiHeadAttention_13529146982602
// MI455X (gfx1250) — matrix operand order repaired and checked by reading, not compiled, not run on hardware
//
#include <hip/hip_runtime.h>

// ---------------------------------------------------------------------------
// CDNA5 (gfx1250) types for WMMA
// ---------------------------------------------------------------------------
typedef __attribute__((ext_vector_type(16))) __bf16 v16bf;
typedef __attribute__((ext_vector_type(8)))  float  v8f;

union BF16Frag {
    v16bf v;
    __bf16 e[16];
    uint4  q[2];      // two 16-byte chunks
};

static __device__ inline __bf16 f2bf(float f) {
    unsigned u = __builtin_bit_cast(unsigned, f);
    unsigned r = u + 0x7FFFu + ((u >> 16) & 1u);   // round-to-nearest-even
    unsigned short h = (unsigned short)(r >> 16);
    return __builtin_bit_cast(__bf16, h);
}

static __device__ inline v8f wmma_bf16(v16bf a, v16bf b, v8f c) {
    // D = A(16x32 bf16) * B(32x16 bf16) + C(16x16 f32)
    return __builtin_amdgcn_wmma_f32_16x16x32_bf16(
        false, a, false, b, (short)0, c, false, false);
}

// A fragment (16x32, MxK) from bf16 row-major src (stride lda elements).
static __device__ inline v16bf load_a_bf16(const __bf16* src, int lda,
                                           int m0, int k0, int lane) {
    int hf = lane >> 4, row = lane & 15;
    const __bf16* p = src + (size_t)(m0 + row) * lda + k0;
    BF16Frag a;
    a.q[0] = *(const uint4*)(p + 8 * hf);
    a.q[1] = *(const uint4*)(p + 16 + 8 * hf);
    return a.v;
}

// B fragment (32x16, KxN) from Bt[n][k] row-major (B column-major), stride ldb.
static __device__ inline v16bf load_b_bf16(const __bf16* bt, int ldb,
                                           int n0, int k0, int lane) {
    int hf = lane >> 4, col = lane & 15;
    const __bf16* p = bt + (size_t)(n0 + col) * ldb + k0;
    BF16Frag b;
    b.q[0] = *(const uint4*)(p + 8 * hf);
    b.q[1] = *(const uint4*)(p + 16 + 8 * hf);
    return b.v;
}

#define V8F_ZERO {0.f,0.f,0.f,0.f,0.f,0.f,0.f,0.f}

// Problem constants
#define BATCH 4
#define TSEQ  2048
#define CDIM  1024
#define HEADS 16
#define DHEAD 64
#define MROWS (BATCH * TSEQ)   // 8192

// ---------------------------------------------------------------------------
// gfx1250 async global->LDS staging (guarded; sync fallback otherwise)
// ---------------------------------------------------------------------------
#if defined(__AMDGCN__) && __has_builtin(__builtin_amdgcn_global_load_async_to_lds_b128)
#define HAVE_ASYNC_LDS 1
#else
#define HAVE_ASYNC_LDS 0
#endif

typedef int v4i_async __attribute__((vector_size(16)));
typedef __attribute__((address_space(1))) v4i_async* gbl_v4p;
typedef __attribute__((address_space(3))) v4i_async* lds_v4p;

static __device__ inline void async_copy_16B(const __bf16* g, __bf16* l) {
#if HAVE_ASYNC_LDS
    __builtin_amdgcn_global_load_async_to_lds_b128(
        (gbl_v4p)(v4i_async*)(void*)const_cast<__bf16*>(g),
        (lds_v4p)(v4i_async*)(void*)l,
        /*offset=*/0, /*cpol=*/0);
#else
    *(uint4*)l = *(const uint4*)g;
#endif
}

#if HAVE_ASYNC_LDS
#if __has_builtin(__builtin_amdgcn_s_wait_asynccnt)
#define WAIT_STAGE(n) do { __builtin_amdgcn_s_wait_asynccnt(n); \
                           asm volatile("" ::: "memory"); } while (0)
#else
#define WAIT_STAGE(n) asm volatile("s_wait_asynccnt %0" :: "i"(n) : "memory")
#endif
#else
#define WAIT_STAGE(n) __syncthreads()
#endif

// ---------------------------------------------------------------------------
// Kernel A: convert Q/K/V activations fp32 -> bf16 (one pass, vectorized)
// ---------------------------------------------------------------------------
__global__ __launch_bounds__(256) void conv_kernel(
    const float* __restrict__ Q, const float* __restrict__ K,
    const float* __restrict__ V,
    __bf16* __restrict__ Qb, __bf16* __restrict__ Kb, __bf16* __restrict__ Vb)
{
    int z = blockIdx.y;
    const float* src = (z == 0) ? Q : (z == 1) ? K : V;
    __bf16*      dst = (z == 0) ? Qb : (z == 1) ? Kb : Vb;
    size_t i = ((size_t)blockIdx.x * 256 + threadIdx.x) * 4;
    float4 f = *(const float4*)(src + i);
    union { __bf16 e[4]; uint2 u; } o;
    o.e[0] = f2bf(f.x); o.e[1] = f2bf(f.y);
    o.e[2] = f2bf(f.z); o.e[3] = f2bf(f.w);
    *(uint2*)(dst + i) = o.u;
}

// ---------------------------------------------------------------------------
// Kernel B: transpose + convert the four 1024x1024 fp32 weights to bf16
// (column-major so B-fragment loads are contiguous)
// ---------------------------------------------------------------------------
__global__ __launch_bounds__(256) void transw_kernel(
    const float* __restrict__ Wq, const float* __restrict__ Wk,
    const float* __restrict__ Wv, const float* __restrict__ Wo,
    __bf16* __restrict__ WqT, __bf16* __restrict__ WkT,
    __bf16* __restrict__ WvT, __bf16* __restrict__ WoT)
{
    __shared__ float tile[32][33];
    int z = blockIdx.z;
    const float* src = (z == 0) ? Wq : (z == 1) ? Wk : (z == 2) ? Wv : Wo;
    __bf16*      dst = (z == 0) ? WqT : (z == 1) ? WkT : (z == 2) ? WvT : WoT;

    int x  = blockIdx.x * 32 + threadIdx.x;
    int y0 = blockIdx.y * 32;
    for (int i = threadIdx.y; i < 32; i += 8)
        tile[i][threadIdx.x] = src[(size_t)(y0 + i) * CDIM + x];
    __syncthreads();
    int xo  = blockIdx.y * 32 + threadIdx.x;
    int yo0 = blockIdx.x * 32;
    for (int i = threadIdx.y; i < 32; i += 8)
        dst[(size_t)(yo0 + i) * CDIM + xo] = f2bf(tile[threadIdx.x][i]);
}

// ---------------------------------------------------------------------------
// Kernel C: fused QKV projection, one wave per 16x64 output strip.
//   mode 0: Qh[b][h][t][d] (pre-scaled by 1/8)   mode 1: Kh[b][h][t][d]
//   mode 2: Vt[b][h][d][t] (transposed for PV B-fragments)
// ---------------------------------------------------------------------------
__global__ __launch_bounds__(128) void qkv_kernel(
    const __bf16* __restrict__ Qb, const __bf16* __restrict__ Kb,
    const __bf16* __restrict__ Vb,
    const __bf16* __restrict__ WqT, const __bf16* __restrict__ WkT,
    const __bf16* __restrict__ WvT,
    const float* __restrict__ bq, const float* __restrict__ bk,
    const float* __restrict__ bv,
    __bf16* __restrict__ Qh, __bf16* __restrict__ Kh,
    __bf16* __restrict__ Vt)
{
    int mode = blockIdx.z;
    const __bf16* X    = (mode == 0) ? Qb  : (mode == 1) ? Kb  : Vb;
    const __bf16* Wt   = (mode == 0) ? WqT : (mode == 1) ? WkT : WvT;
    const float*  bias = (mode == 0) ? bq  : (mode == 1) ? bk  : bv;

    int lane = threadIdx.x & 31;
    int wave = threadIdx.x >> 5;
    int n0 = (blockIdx.x * 4 + wave) * 64;
    int m0 = blockIdx.y * 16;

    v8f acc[4] = {V8F_ZERO, V8F_ZERO, V8F_ZERO, V8F_ZERO};
    for (int k0 = 0; k0 < CDIM; k0 += 32) {
        __builtin_prefetch(X + (size_t)(m0 + (lane & 15)) * CDIM + k0 + 128, 0, 1);
        v16bf a = load_a_bf16(X, CDIM, m0, k0, lane);   // reused 4x
#pragma unroll
        for (int j = 0; j < 4; ++j)
            acc[j] = wmma_bf16(a, load_b_bf16(Wt, CDIM, n0 + 16 * j, k0, lane),
                               acc[j]);
    }

    int hf = lane >> 4, col = lane & 15;
#pragma unroll
    for (int j = 0; j < 4; ++j) {
        int N = n0 + 16 * j + col;
        float bval = bias[N];
        int h = N >> 6, d = N & 63;
#pragma unroll
        for (int r = 0; r < 8; ++r) {
            int M = m0 + r + 8 * hf;
            int bb = M >> 11, t = M & (TSEQ - 1);
            float val = acc[j][r] + bval;
            if (mode == 0) val *= 0.125f;   // 1/sqrt(DK) folded into Q
            __bf16 o = f2bf(val);
            if (mode == 2) {
                Vt[(((size_t)bb * HEADS + h) * DHEAD + d) * TSEQ + t] = o;
            } else {
                __bf16* dst = (mode == 0) ? Qh : Kh;
                dst[(((size_t)bb * HEADS + h) * TSEQ + t) * DHEAD + d] = o;
            }
        }
    }
}

// ---------------------------------------------------------------------------
// Kernel D: causal flash attention, one wave per (b, h, 16-query tile).
// K/V 32-key tiles double-buffered in LDS via async global->LDS copies.
// ---------------------------------------------------------------------------
#define PSTR 40   // LDS P-tile row stride (bf16), keeps 16B-aligned chunks

__global__ __launch_bounds__(32) void attn_kernel(
    const __bf16* __restrict__ Qh, const __bf16* __restrict__ Kh,
    const __bf16* __restrict__ Vt, __bf16* __restrict__ Attn)
{
    __shared__ __align__(16) __bf16 Pl[16 * PSTR];
    __shared__ __align__(16) __bf16 Ktile[2][32 * DHEAD];   // [key][d]
    __shared__ __align__(16) __bf16 Vtile[2][DHEAD * 32];   // [d][key]

    int lane = threadIdx.x & 31;
    int qt = blockIdx.x, h = blockIdx.y, b = blockIdx.z;
    int hf = lane >> 4, col = lane & 15;

    const __bf16* Qp = Qh + ((size_t)b * HEADS + h) * TSEQ * DHEAD;
    const __bf16* Kp = Kh + ((size_t)b * HEADS + h) * TSEQ * DHEAD;
    const __bf16* Vp = Vt + ((size_t)b * HEADS + h) * DHEAD * TSEQ;
    int q0 = qt * 16;

    v16bf aq0 = load_a_bf16(Qp, DHEAD, q0, 0,  lane);
    v16bf aq1 = load_a_bf16(Qp, DHEAD, q0, 32, lane);

    v8f o0 = V8F_ZERO, o1 = V8F_ZERO, o2 = V8F_ZERO, o3 = V8F_ZERO;
    float m_run[8], l_run[8];
#pragma unroll
    for (int r = 0; r < 8; ++r) { m_run[r] = -1e30f; l_run[r] = 0.f; }

    int kblocks = (q0 + 15) / 32 + 1;   // causal: keys <= q0+15

    // ---- stage block 0 (16 async b128 per block: 8 for K, 8 for V)
    {
#pragma unroll
        for (int i = 0; i < 8; ++i) {
            int c = i * 32 + lane;
            async_copy_16B(Kp + c * 8, &Ktile[0][c * 8]);
        }
#pragma unroll
        for (int i = 0; i < 8; ++i) {
            int c = i * 32 + lane;
            int row = c >> 2, sub = c & 3;
            async_copy_16B(Vp + (size_t)row * TSEQ + sub * 8,
                           &Vtile[0][row * 32 + sub * 8]);
        }
    }

    for (int kb = 0; kb < kblocks; ++kb) {
        int kb0 = kb * 32;
        int cur = kb & 1;
        bool more = (kb + 1) < kblocks;

        // ---- prefetch next key block into the other buffer
        if (more) {
            int nb0 = kb0 + 32;
#pragma unroll
            for (int i = 0; i < 8; ++i) {
                int c = i * 32 + lane;
                async_copy_16B(Kp + (size_t)nb0 * DHEAD + c * 8,
                               &Ktile[cur ^ 1][c * 8]);
            }
#pragma unroll
            for (int i = 0; i < 8; ++i) {
                int c = i * 32 + lane;
                int row = c >> 2, sub = c & 3;
                async_copy_16B(Vp + (size_t)row * TSEQ + nb0 + sub * 8,
                               &Vtile[cur ^ 1][row * 32 + sub * 8]);
            }
        }
        if (more) WAIT_STAGE(16);   // current block's 16 loads retired
        else      WAIT_STAGE(0);

        // ---- S = Q * K^T from LDS K tile
        v8f s0 = V8F_ZERO, s1 = V8F_ZERO;
        s0 = wmma_bf16(aq0, load_b_bf16(Ktile[cur], DHEAD, 0,  0,  lane), s0);
        s0 = wmma_bf16(aq1, load_b_bf16(Ktile[cur], DHEAD, 0,  32, lane), s0);
        s1 = wmma_bf16(aq0, load_b_bf16(Ktile[cur], DHEAD, 16, 0,  lane), s1);
        s1 = wmma_bf16(aq1, load_b_bf16(Ktile[cur], DHEAD, 16, 32, lane), s1);

        // ---- causal mask (only the diagonal-straddling final block)
        if (kb == kblocks - 1) {
#pragma unroll
            for (int r = 0; r < 8; ++r) {
                int q = q0 + r + 8 * hf;
                if (kb0 + col      > q) s0[r] = -1e30f;
                if (kb0 + 16 + col > q) s1[r] = -1e30f;
            }
        }

        // ---- online softmax (rows live across the 16 lanes of each half)
        float p0[8], p1[8], fac[8];
#pragma unroll
        for (int r = 0; r < 8; ++r) {
            float mr = fmaxf(s0[r], s1[r]);
            mr = fmaxf(mr, __shfl_xor(mr, 1, 32));
            mr = fmaxf(mr, __shfl_xor(mr, 2, 32));
            mr = fmaxf(mr, __shfl_xor(mr, 4, 32));
            mr = fmaxf(mr, __shfl_xor(mr, 8, 32));
            float mnew = fmaxf(m_run[r], mr);
            fac[r] = __expf(m_run[r] - mnew);
            m_run[r] = mnew;
            float e0 = __expf(s0[r] - mnew);
            float e1 = __expf(s1[r] - mnew);
            p0[r] = e0; p1[r] = e1;
            float rs = e0 + e1;
            rs += __shfl_xor(rs, 1, 32);
            rs += __shfl_xor(rs, 2, 32);
            rs += __shfl_xor(rs, 4, 32);
            rs += __shfl_xor(rs, 8, 32);
            l_run[r] = l_run[r] * fac[r] + rs;
        }
#pragma unroll
        for (int r = 0; r < 8; ++r) {
            o0[r] *= fac[r]; o1[r] *= fac[r];
            o2[r] *= fac[r]; o3[r] *= fac[r];
        }

        // ---- P: C-layout f32 -> A-layout bf16 via LDS round trip
        __syncthreads();
#pragma unroll
        for (int r = 0; r < 8; ++r) {
            Pl[(r + 8 * hf) * PSTR + col]      = f2bf(p0[r]);
            Pl[(r + 8 * hf) * PSTR + 16 + col] = f2bf(p1[r]);
        }
        __syncthreads();
        v16bf ap = load_a_bf16(Pl, PSTR, 0, 0, lane);

        // ---- O += P * V from LDS V tile (Vtile[d][key_local], stride 32)
        o0 = wmma_bf16(ap, load_b_bf16(Vtile[cur], 32, 0,  0, lane), o0);
        o1 = wmma_bf16(ap, load_b_bf16(Vtile[cur], 32, 16, 0, lane), o1);
        o2 = wmma_bf16(ap, load_b_bf16(Vtile[cur], 32, 32, 0, lane), o2);
        o3 = wmma_bf16(ap, load_b_bf16(Vtile[cur], 32, 48, 0, lane), o3);
    }

    // ---- epilogue: normalize, write Attn[b][t][h*64+d] bf16
#pragma unroll
    for (int r = 0; r < 8; ++r) {
        float inv = 1.0f / l_run[r];
        int t = q0 + r + 8 * hf;
        size_t base = ((size_t)b * TSEQ + t) * CDIM + h * DHEAD;
        Attn[base +      col] = f2bf(o0[r] * inv);
        Attn[base + 16 + col] = f2bf(o1[r] * inv);
        Attn[base + 32 + col] = f2bf(o2[r] * inv);
        Attn[base + 48 + col] = f2bf(o3[r] * inv);
    }
}

// ---------------------------------------------------------------------------
// Kernel E: output projection  Out = Attn * Wo + bo  (fp32 out), 16x64/wave
// ---------------------------------------------------------------------------
__global__ __launch_bounds__(128) void oproj_kernel(
    const __bf16* __restrict__ Attn, const __bf16* __restrict__ WoT,
    const float* __restrict__ bo, float* __restrict__ Out)
{
    int lane = threadIdx.x & 31;
    int wave = threadIdx.x >> 5;
    int n0 = (blockIdx.x * 4 + wave) * 64;
    int m0 = blockIdx.y * 16;

    v8f acc[4] = {V8F_ZERO, V8F_ZERO, V8F_ZERO, V8F_ZERO};
    for (int k0 = 0; k0 < CDIM; k0 += 32) {
        v16bf a = load_a_bf16(Attn, CDIM, m0, k0, lane);
#pragma unroll
        for (int j = 0; j < 4; ++j)
            acc[j] = wmma_bf16(a, load_b_bf16(WoT, CDIM, n0 + 16 * j, k0, lane),
                               acc[j]);
    }

    int hf = lane >> 4, col = lane & 15;
#pragma unroll
    for (int j = 0; j < 4; ++j) {
        float bval = bo[n0 + 16 * j + col];
#pragma unroll
        for (int r = 0; r < 8; ++r)
            Out[(size_t)(m0 + r + 8 * hf) * CDIM + n0 + 16 * j + col] =
                acc[j][r] + bval;
    }
}

// ---------------------------------------------------------------------------
// Host launcher
// ---------------------------------------------------------------------------
extern "C" void kernel_launch(void* const* d_in, const int* in_sizes, int n_in,
                              void* d_out, int out_size, void* d_ws, size_t ws_size,
                              hipStream_t stream) {
    (void)in_sizes; (void)n_in; (void)out_size; (void)ws_size;

    const float* Q  = (const float*)d_in[0];
    const float* K  = (const float*)d_in[1];
    const float* V  = (const float*)d_in[2];
    /* d_in[3] = mask: causal, reproduced analytically in attn_kernel */
    const float* Wq = (const float*)d_in[4];
    const float* bq = (const float*)d_in[5];
    const float* Wk = (const float*)d_in[6];
    const float* bk = (const float*)d_in[7];
    const float* Wv = (const float*)d_in[8];
    const float* bv = (const float*)d_in[9];
    const float* Wo = (const float*)d_in[10];
    const float* bo = (const float*)d_in[11];
    float* Out = (float*)d_out;

    char* ws = (char*)d_ws;
    const size_t WBYTES = (size_t)CDIM * CDIM * sizeof(__bf16);                  // 2 MB
    const size_t XBYTES = (size_t)MROWS * CDIM * sizeof(__bf16);                 // 16 MB
    const size_t HBYTES = (size_t)BATCH * HEADS * TSEQ * DHEAD * sizeof(__bf16); // 16 MB
    size_t off = 0;
    __bf16* WqT = (__bf16*)(ws + off); off += WBYTES;
    __bf16* WkT = (__bf16*)(ws + off); off += WBYTES;
    __bf16* WvT = (__bf16*)(ws + off); off += WBYTES;
    __bf16* WoT = (__bf16*)(ws + off); off += WBYTES;
    __bf16* Qb  = (__bf16*)(ws + off); off += XBYTES;
    __bf16* Kb  = (__bf16*)(ws + off); off += XBYTES;
    __bf16* Vb  = (__bf16*)(ws + off); off += XBYTES;
    __bf16* Qh  = (__bf16*)(ws + off); off += HBYTES;
    __bf16* Kh  = (__bf16*)(ws + off); off += HBYTES;
    __bf16* Vt  = (__bf16*)(ws + off); off += HBYTES;
    __bf16* At  = (__bf16*)(ws + off); off += HBYTES;

    // A) activations fp32 -> bf16 (one pass)
    conv_kernel<<<dim3(MROWS * CDIM / 4 / 256, 3), dim3(256), 0, stream>>>(
        Q, K, V, Qb, Kb, Vb);

    // B) weights: transpose + convert to bf16
    transw_kernel<<<dim3(CDIM / 32, CDIM / 32, 4), dim3(32, 8), 0, stream>>>(
        Wq, Wk, Wv, Wo, WqT, WkT, WvT, WoT);

    // C) QKV projections (z selects Q/K/V), 16x64 per wave
    qkv_kernel<<<dim3(CDIM / 256, MROWS / 16, 3), dim3(128), 0, stream>>>(
        Qb, Kb, Vb, WqT, WkT, WvT, bq, bk, bv, Qh, Kh, Vt);

    // D) causal flash attention with async-LDS double buffering
    attn_kernel<<<dim3(TSEQ / 16, HEADS, BATCH), dim3(32), 0, stream>>>(
        Qh, Kh, Vt, At);

    // E) output projection
    oproj_kernel<<<dim3(CDIM / 256, MROWS / 16), dim3(128), 0, stream>>>(
        At, WoT, bo, Out);
}
